// MeshGraphNet_11527692222557
// MI455X (gfx1250) — compile-verified, matrix operand order checked by reading, not run on hardware
//
#include <hip/hip_runtime.h>
#include <hip/hip_bf16.h>
#include <stdint.h>

#define NNODES 20000
#define NEDGES 120000
#define HID 128
#define NLAYERS 6

typedef _Float16 v16h __attribute__((ext_vector_type(16)));
typedef _Float16 v4h  __attribute__((ext_vector_type(4)));
typedef float    v8f  __attribute__((ext_vector_type(8)));
typedef unsigned int u32x4 __attribute__((ext_vector_type(4)));
typedef int      i32x8 __attribute__((ext_vector_type(8)));
typedef int      i32x4 __attribute__((ext_vector_type(4)));

// ---------------- LDS layout (static, 55,296 B -> 5 WGs/WGP of 320KB) ----------
#define SA_OFF    0        // 64 x (KC+8) f16 A staging (max 17408 B)
#define SH0_OFF   17408    // 64 x 136 f16 stage-1 activations
#define SH1_OFF   34816    // 64 x 136 f16 stage-2 activations
#define SSTAT_OFF 52224    // 64 x {mu, rsigma}
#define SPAR_OFF  52736    // 5 x 128 f32 biases + LN params (TDM-staged)
#define SMEM_BYTES 55296   // sY (64x128 f32) aliases [0, 32768) = sA+sH0

struct SegDesc { const float* src; const int* gather; int K; };
struct MlpArgs {
  SegDesc seg[3];
  const _Float16* W0; const _Float16* W1; const _Float16* W2;  // packed f16 B-tiles
  const float* b0; const float* b1; const float* b2;           // f32 biases (TDM src)
  const float* ln_g; const float* ln_b;
  const float* resid;
  float* outF;           // mode 0/1 f32 output (HID wide)
  float* scatterDst;     // aggF or null (mode 1)
  const int* scatterIdx; // dst indices (mode 1 edge)
  float* out3;           // mode 2 decoder output
  int M;                 // total rows
  int n3real;            // 128 or 3
  int mode;              // 0 = plain, 1 = LN+residual(+scatter), 2 = decoder
};

// ---------------- TDM: 1-D async copy global -> LDS (CDNA5 tensor data mover) ---
// amdgpu-toolchain (clang-23) 6-arg form:
//   (uint32x4 g0, int32x8 g1, int32x4 g2, int32x4 g3, int32x8, i32 cpol)
__device__ __forceinline__ void tdm_load_1d(const void* gsrc, unsigned lds_off,
                                            unsigned nelem, unsigned size_log2) {
  unsigned long long ga = (unsigned long long)(uintptr_t)gsrc;
  u32x4 g0;
  g0.x = 1u;                                                // count=1, user descriptor
  g0.y = lds_off;                                           // D#.lds_addr
  g0.z = (unsigned)ga;                                      // global_addr[31:0]
  g0.w = (unsigned)((ga >> 32) & 0x01ffffffu) | (2u << 30); // addr[56:32] | type=2
  i32x8 g1;
  g1[0] = (int)(size_log2 << 16);               // data_size (0=1B..3=8B), no multicast
  g1[1] = (int)((nelem & 0xffffu) << 16);       // tensor_dim0[15:0]
  g1[2] = (int)((nelem >> 16) & 0xffffu);       // tensor_dim0[31:16], tensor_dim1=0
  g1[3] = (int)((nelem & 0xffffu) << 16);       // tile_dim0
  g1[4] = 0;                                    // tile_dim1/2 unused
  g1[5] = (int)nelem;                           // tensor_dim0_stride
  g1[6] = 0; g1[7] = 0;
  i32x4 z4; z4[0] = 0; z4[1] = 0; z4[2] = 0; z4[3] = 0;
  i32x8 z8;
#pragma unroll
  for (int i = 0; i < 8; ++i) z8[i] = 0;
  __builtin_amdgcn_tensor_load_to_lds(g0, g1, z4, z4, z8, 0);
}

// ---------------- fragment loads (ISA 16-bit A/B layouts) -----------------------
// A: lanes 0-15 row M=lane K{0..7,16..23}; lanes 16-31 row M=lane-16 K{8..15,24..31}
__device__ __forceinline__ v16h load_afrag(const _Float16* ap) {
  v16h af;
  ((uint4*)&af)[0] = *(const uint4*)(ap);
  ((uint4*)&af)[1] = *(const uint4*)(ap + 16);
  return af;
}
__device__ __forceinline__ v16h load_bfrag(const _Float16* tileBase, int lane) {
  v16h bf;
  const uint4* q = (const uint4*)(tileBase) + (lane << 1);
  ((uint4*)&bf)[0] = q[0];
  ((uint4*)&bf)[1] = q[1];
  return bf;
}

// ---------------- wave GEMM: acc[4] += A(16 x 32*KT) * B(32*KT x 64) ------------
// Straight-line, fully unrolled, no predication: every wave owns 4 N-tiles.
template <int KT>
__device__ __forceinline__ void gemm_block(const _Float16* sAp, int lda,
                                           const _Float16* B, v8f* acc,
                                           int mstrip, int nhalf, int lane, int base1) {
  const _Float16* arow = sAp + (mstrip * 16 + (lane & 15)) * lda + base1;
#pragma unroll
  for (int kt = 0; kt < KT; ++kt) {
    v16h af = load_afrag(arow + kt * 32);
#pragma unroll
    for (int nt = 0; nt < 4; ++nt) {
      v16h bf = load_bfrag(B + ((size_t)(kt * 8 + nhalf * 4 + nt) << 9), lane);
      acc[nt] = __builtin_amdgcn_wmma_f32_16x16x32_f16(
          false, af, false, bf, (short)0, acc[nt], false, false);
    }
  }
}

// Decoder stage 3: single padded N-tile (packed W2 has tile stride 1)
__device__ __forceinline__ void gemm_1tile(const _Float16* sAp, int lda,
                                           const _Float16* B, v8f& acc,
                                           int mstrip, int lane, int base1) {
  const _Float16* arow = sAp + (mstrip * 16 + (lane & 15)) * lda + base1;
#pragma unroll
  for (int kt = 0; kt < 4; ++kt) {
    v16h af = load_afrag(arow + kt * 32);
    v16h bf = load_bfrag(B + ((size_t)kt << 9), lane);
    acc = __builtin_amdgcn_wmma_f32_16x16x32_f16(
        false, af, false, bf, (short)0, acc, false, false);
  }
}

// C-tile (VGPR r: M=r lanes0-15, M=8+r lanes16-31; N=lane&15) -> LDS f16 w/ bias+ReLU
__device__ __forceinline__ void frag_store_f16(const v8f* acc, _Float16* sOut, int ldo,
                                               const float* bias, int mstrip, int nhalf,
                                               int lane, bool relu) {
  int colbase = nhalf * 64 + (lane & 15);
  int rowbase = mstrip * 16 + ((lane & 16) ? 8 : 0);
#pragma unroll
  for (int nt = 0; nt < 4; ++nt) {
    int col = colbase + nt * 16;
    float bv = bias[col];
#pragma unroll
    for (int r = 0; r < 8; ++r) {
      float v = acc[nt][r] + bv;
      if (relu) v = fmaxf(v, 0.f);
      sOut[(rowbase + r) * ldo + col] = (_Float16)v;
    }
  }
}

// ---------------- fused 3-layer MLP (+LN+residual+scatter) ----------------------
// Specialized on (NSEG, KC): stage 1 is straight-line, no acc phis.
template <int NSEG, int KC>
__global__ __launch_bounds__(256) void fused_mlp3(MlpArgs a) {
  __shared__ __align__(16) unsigned char smem[SMEM_BYTES];
  _Float16* sA    = (_Float16*)(smem + SA_OFF);
  _Float16* sH0   = (_Float16*)(smem + SH0_OFF);
  _Float16* sH1   = (_Float16*)(smem + SH1_OFF);
  float*    sY    = (float*)(smem);           // aliases sA/sH0 (dead by stage 3)
  float*    sStat = (float*)(smem + SSTAT_OFF);
  float*    sPar  = (float*)(smem + SPAR_OFF);

  const int tid  = threadIdx.x;
  const int lane = tid & 31;
  // provably wave-uniform (SGPR) wave coordinates -> scalar branches, scalar addrs
  const int wave   = __builtin_amdgcn_readfirstlane(tid >> 5);
  const int mstrip = wave & 3;
  const int nhalf  = wave >> 2;
  const int base1  = (lane & 16) ? 8 : 0;
  const int tileM0 = blockIdx.x * 64;

  // async-stage biases / LN params into LDS via the tensor data mover
  if (wave == 0) {
    unsigned spar = (unsigned)(uintptr_t)smem + SPAR_OFF;
    tdm_load_1d(a.b0, spar + 0u * 512u, HID, 2);
    tdm_load_1d(a.b1, spar + 1u * 512u, HID, 2);
    tdm_load_1d(a.b2, spar + 2u * 512u, (unsigned)a.n3real, 2);
    if (a.mode == 1) {
      tdm_load_1d(a.ln_g, spar + 3u * 512u, HID, 2);
      tdm_load_1d(a.ln_b, spar + 4u * 512u, HID, 2);
    }
  }

  v8f acc[4];
  v8f vzero = {};
#pragma unroll
  for (int t = 0; t < 4; ++t) acc[t] = vzero;

  // ---- stage 1: K-chunked over input segments (gather + f32->f16 into LDS) ----
  constexpr int LDA = KC + 8;                  // +16B pad breaks LDS bank conflicts
#pragma unroll
  for (int s = 0; s < NSEG; ++s) {
    const int segK = a.seg[s].K;
    const int chunks = segK >> 2;
    const float* src = a.seg[s].src;
    const int* gat = a.seg[s].gather;
    for (int i = tid; i < 64 * chunks; i += 256) {
      int r = i / chunks, c = i - r * chunks;
      int grow = tileM0 + r; if (grow >= a.M) grow = a.M - 1;
      int srow = gat ? gat[grow] : grow;
      float4 f = *(const float4*)(src + (size_t)srow * segK + (c << 2));
      v4h h; h.x = (_Float16)f.x; h.y = (_Float16)f.y;
             h.z = (_Float16)f.z; h.w = (_Float16)f.w;
      *(v4h*)(sA + r * LDA + (c << 2)) = h;
    }
    const int rem = KC - segK;                 // zero-pad K for the encoders
    if (rem > 0) {
      for (int i = tid; i < 64 * rem; i += 256) {
        int r = i / rem, c = i - r * rem;
        sA[r * LDA + segK + c] = (_Float16)0.f;
      }
    }
    __syncthreads();
    gemm_block<KC / 32>(sA, LDA, a.W0 + (size_t)s * (KC / 32) * 8 * 512,
                        acc, mstrip, nhalf, lane, base1);
    __syncthreads();
  }

  if (wave == 0) __builtin_amdgcn_s_wait_tensorcnt(0);  // biases are in LDS now
  __syncthreads();
  frag_store_f16(acc, sH0, 136, sPar + 0 * 128, mstrip, nhalf, lane, true);
  __syncthreads();

  // ---- stage 2: [64,128] x [128,128] ----
#pragma unroll
  for (int t = 0; t < 4; ++t) acc[t] = vzero;
  gemm_block<4>(sH0, 136, a.W1, acc, mstrip, nhalf, lane, base1);
  frag_store_f16(acc, sH1, 136, sPar + 1 * 128, mstrip, nhalf, lane, true);
  __syncthreads();

  // ---- stage 3: [64,128] x [128,N3] ----
  const int rowbase = mstrip * 16 + ((lane & 16) ? 8 : 0);
  const int colbase = nhalf * 64 + (lane & 15);
  const float* bias2 = sPar + 2 * 128;

  if (a.mode == 2) {                               // decoder: N=3 of padded 16
    if (nhalf == 0) {
      v8f dacc = vzero;
      gemm_1tile(sH1, 136, a.W2, dacc, mstrip, lane, base1);
      int col = lane & 15;
      if (col < a.n3real) {
        float bv = bias2[col];
#pragma unroll
        for (int r = 0; r < 8; ++r) {
          int grow = tileM0 + rowbase + r;
          if (grow < a.M) a.out3[(size_t)grow * a.n3real + col] = dacc[r] + bv;
        }
      }
    }
    return;
  }

#pragma unroll
  for (int t = 0; t < 4; ++t) acc[t] = vzero;
  gemm_block<4>(sH1, 136, a.W2, acc, mstrip, nhalf, lane, base1);

  if (a.mode == 0) {                               // encoders: plain f32 store
#pragma unroll
    for (int nt = 0; nt < 4; ++nt) {
      int col = colbase + nt * 16;
      float bv = bias2[col];
#pragma unroll
      for (int r = 0; r < 8; ++r) {
        int grow = tileM0 + rowbase + r;
        if (grow < a.M) a.outF[(size_t)grow * HID + col] = acc[nt][r] + bv;
      }
    }
  } else {                                         // mode 1: LN + residual (+scatter)
#pragma unroll
    for (int nt = 0; nt < 4; ++nt) {
      int col = colbase + nt * 16;
      float bv = bias2[col];
#pragma unroll
      for (int r = 0; r < 8; ++r)
        sY[(rowbase + r) * HID + col] = acc[nt][r] + bv;
    }
    __syncthreads();
    if (tid < 64) {
      const float* yr = sY + tid * HID;
      float s = 0.f, sq = 0.f;
      for (int c = 0; c < HID; ++c) { float x = yr[c]; s += x; sq += x * x; }
      float mu = s * (1.f / HID);
      float var = sq * (1.f / HID) - mu * mu;
      sStat[2 * tid]     = mu;
      sStat[2 * tid + 1] = rsqrtf(var + 1e-5f);
    }
    __syncthreads();
    const float* g  = sPar + 3 * 128;
    const float* bb = sPar + 4 * 128;
    for (int i = tid; i < 64 * HID; i += 256) {
      int r = i >> 7, c = i & (HID - 1);
      int grow = tileM0 + r;
      if (grow < a.M) {
        float ln = (sY[i] - sStat[2 * r]) * sStat[2 * r + 1] * g[c] + bb[c];
        float ov = a.resid[(size_t)grow * HID + c] + ln;
        a.outF[(size_t)grow * HID + c] = ov;
        if (a.scatterDst) {                        // fused segment_sum(e, dst)
          int gi = a.scatterIdx[grow];
          __hip_atomic_fetch_add(&a.scatterDst[(size_t)gi * HID + c], ov,
                                 __ATOMIC_RELAXED, __HIP_MEMORY_SCOPE_AGENT);
        }
      }
    }
  }
}

// ---- pack f32 W[K,N] -> f16 WMMA B fragments (32x16 tiles, per-lane contiguous) --
__global__ __launch_bounds__(256) void pack_w(const float* __restrict__ W,
                                              _Float16* __restrict__ out,
                                              int Kr, int Nr, int Kp, int Np) {
  int NT = Np >> 4;
  int total = Kp * Np;
  for (int i = blockIdx.x * blockDim.x + threadIdx.x; i < total;
       i += gridDim.x * blockDim.x) {
    int h = i & 15;
    int l = (i >> 4) & 31;
    int tile = i >> 9;
    int kt = tile / NT;
    int nt = tile - kt * NT;
    int k = kt * 32 + ((l & 16) ? 8 : 0) + ((h < 8) ? h : h + 8);
    int n = nt * 16 + (l & 15);
    float v = (k < Kr && n < Nr) ? W[(size_t)k * Nr + n] : 0.f;
    out[i] = (_Float16)v;
  }
}

__global__ __launch_bounds__(256) void zero_f32(float* p, long long n) {
  long long i = (long long)blockIdx.x * blockDim.x + threadIdx.x;
  long long stride = (long long)gridDim.x * blockDim.x;
  for (; i < n; i += stride) p[i] = 0.f;
}

// --------------------------------- host driver ----------------------------------
extern "C" void kernel_launch(void* const* d_in, const int* in_sizes, int n_in,
                              void* d_out, int out_size, void* d_ws, size_t ws_size,
                              hipStream_t stream) {
  (void)in_sizes; (void)out_size; (void)ws_size;
  if (n_in < 117) return;

  const float* node_feat = (const float*)d_in[0];
  const float* edge_feat = (const float*)d_in[1];
  const int*   eidx      = (const int*)d_in[2];
  const int*   srcI = eidx;
  const int*   dstI = eidx + NEDGES;
  auto P = [&](int i) { return (const float*)d_in[i]; };
  // d_in: 3..8 node_enc (W,b)x3 ; 9..14 edge_enc ; 15..20 decoder ;
  //       21+8l edge_blocks {W0,b0,W1,b1,W2,b2,g,b} ; 69+8l node_blocks

  // workspace carve-out
  char* base = (char*)d_ws;
  size_t off = 0;
  auto alloc = [&](size_t bytes) -> char* {
    size_t s = (off + 255) & ~(size_t)255;
    off = s + bytes;
    return base + s;
  };
  float* vF   = (float*)alloc((size_t)NNODES * HID * 4);
  float* eF   = (float*)alloc((size_t)NEDGES * HID * 4);
  float* aggF = (float*)alloc((size_t)NNODES * HID * 4);
  auto allocW = [&](int Kp, int Np) { return (_Float16*)alloc((size_t)2 * Kp * Np); };

  _Float16 *neW[3], *eeW[3], *deW[3], *ebW[NLAYERS][3], *nbW[NLAYERS][3];
  neW[0] = allocW(32, 128);  neW[1] = allocW(128, 128); neW[2] = allocW(128, 128);
  eeW[0] = allocW(32, 128);  eeW[1] = allocW(128, 128); eeW[2] = allocW(128, 128);
  deW[0] = allocW(128, 128); deW[1] = allocW(128, 128); deW[2] = allocW(128, 16);
  for (int l = 0; l < NLAYERS; ++l) {
    ebW[l][0] = allocW(384, 128); ebW[l][1] = allocW(128, 128); ebW[l][2] = allocW(128, 128);
    nbW[l][0] = allocW(256, 128); nbW[l][1] = allocW(128, 128); nbW[l][2] = allocW(128, 128);
  }

  auto packL = [&](const float* w, _Float16* d, int Kr, int Nr, int Kp, int Np) {
    int total = Kp * Np;
    pack_w<<<(total + 255) / 256, 256, 0, stream>>>(w, d, Kr, Nr, Kp, Np);
  };
  packL(P(3),  neW[0], 16, 128, 32, 128);
  packL(P(5),  neW[1], 128, 128, 128, 128);
  packL(P(7),  neW[2], 128, 128, 128, 128);
  packL(P(9),  eeW[0], 8, 128, 32, 128);
  packL(P(11), eeW[1], 128, 128, 128, 128);
  packL(P(13), eeW[2], 128, 128, 128, 128);
  packL(P(15), deW[0], 128, 128, 128, 128);
  packL(P(17), deW[1], 128, 128, 128, 128);
  packL(P(19), deW[2], 128, 3, 128, 16);
  for (int l = 0; l < NLAYERS; ++l) {
    int b = 21 + 8 * l, c = 69 + 8 * l;
    packL(P(b + 0), ebW[l][0], 384, 128, 384, 128);
    packL(P(b + 2), ebW[l][1], 128, 128, 128, 128);
    packL(P(b + 4), ebW[l][2], 128, 128, 128, 128);
    packL(P(c + 0), nbW[l][0], 256, 128, 256, 128);
    packL(P(c + 2), nbW[l][1], 128, 128, 128, 128);
    packL(P(c + 4), nbW[l][2], 128, 128, 128, 128);
  }

  const int GN = (NNODES + 63) / 64;   // 313
  const int GE = (NEDGES + 63) / 64;   // 1875

  // encoders (NSEG=1, KC=32)
  MlpArgs a{};
  a.seg[0] = {node_feat, nullptr, 16};
  a.W0 = neW[0]; a.W1 = neW[1]; a.W2 = neW[2];
  a.b0 = P(4); a.b1 = P(6); a.b2 = P(8);
  a.ln_g = P(8); a.ln_b = P(8);
  a.outF = vF; a.M = NNODES; a.n3real = HID; a.mode = 0;
  fused_mlp3<1, 32><<<GN, 256, 0, stream>>>(a);

  a.seg[0] = {edge_feat, nullptr, 8};
  a.W0 = eeW[0]; a.W1 = eeW[1]; a.W2 = eeW[2];
  a.b0 = P(10); a.b1 = P(12); a.b2 = P(14);
  a.outF = eF; a.M = NEDGES;
  fused_mlp3<1, 32><<<GE, 256, 0, stream>>>(a);

  // message-passing layers
  for (int l = 0; l < NLAYERS; ++l) {
    int b = 21 + 8 * l, c = 69 + 8 * l;
    zero_f32<<<512, 256, 0, stream>>>(aggF, (long long)NNODES * HID);

    MlpArgs e{};
    e.seg[0] = {vF, srcI, HID}; e.seg[1] = {vF, dstI, HID}; e.seg[2] = {eF, nullptr, HID};
    e.W0 = ebW[l][0]; e.W1 = ebW[l][1]; e.W2 = ebW[l][2];
    e.b0 = P(b + 1); e.b1 = P(b + 3); e.b2 = P(b + 5);
    e.ln_g = P(b + 6); e.ln_b = P(b + 7);
    e.resid = eF; e.outF = eF; e.scatterDst = aggF; e.scatterIdx = dstI;
    e.M = NEDGES; e.n3real = HID; e.mode = 1;
    fused_mlp3<3, 128><<<GE, 256, 0, stream>>>(e);

    MlpArgs nv{};
    nv.seg[0] = {vF, nullptr, HID}; nv.seg[1] = {aggF, nullptr, HID};
    nv.W0 = nbW[l][0]; nv.W1 = nbW[l][1]; nv.W2 = nbW[l][2];
    nv.b0 = P(c + 1); nv.b1 = P(c + 3); nv.b2 = P(c + 5);
    nv.ln_g = P(c + 6); nv.ln_b = P(c + 7);
    nv.resid = vF; nv.outF = vF;
    nv.M = NNODES; nv.n3real = HID; nv.mode = 1;
    fused_mlp3<2, 128><<<GN, 256, 0, stream>>>(nv);
  }

  // decoder (NSEG=1, KC=128)
  MlpArgs d{};
  d.seg[0] = {vF, nullptr, HID};
  d.W0 = deW[0]; d.W1 = deW[1]; d.W2 = deW[2];
  d.b0 = P(16); d.b1 = P(18); d.b2 = P(20);
  d.ln_g = P(20); d.ln_b = P(20);
  d.out3 = (float*)d_out;
  d.M = NNODES; d.n3real = 3; d.mode = 2;
  fused_mlp3<1, 128><<<GN, 256, 0, stream>>>(d);
}
